// LineGCN2_41712722378984
// MI455X (gfx1250) — matrix operand order repaired and checked by reading, not compiled, not run on hardware
//
#include <hip/hip_runtime.h>
#include <hip/hip_bf16.h>

typedef _Float16 f16;
typedef __attribute__((ext_vector_type(16))) _Float16 v16h;
typedef __attribute__((ext_vector_type(8)))  _Float16 v8h;
typedef __attribute__((ext_vector_type(8)))  float    v8f;
typedef __attribute__((ext_vector_type(4)))  unsigned int u32x4;
typedef __attribute__((ext_vector_type(8)))  int      i32x8;
typedef __attribute__((ext_vector_type(4)))  int      i32x4;

#define HIDF 256   // hidden width == GEMM N
#define TSTR 40    // padded LDS K-stride in halfs (80B: 16B-aligned, conflict-mitigating)

#if defined(__gfx1250__) && __has_builtin(__builtin_amdgcn_tensor_load_to_lds)
#define HAVE_TDM 1
#else
#define HAVE_TDM 0
#endif

// ---------------------------------------------------------------------------
// TDM helper: 2D f16 tile load Global->LDS with padded LDS rows.
// D# built per CDNA5 ISA ch.8: group0 = {flags, lds_addr, global_addr, type=2},
// group1 = {data_size=2B, pad(16DW interval, 4DW amount), dims/strides}.
// Pad turns a 32-half tile row into a 40-half LDS stride "for free".
// This toolchain's builtin is the 6-arg form: (g0, g1, g2, g3, g4, cpol);
// groups 2/3 and the trailing i32x8 are zero (2D tensor, no extra descriptor).
// ---------------------------------------------------------------------------
#if HAVE_TDM
__device__ __forceinline__ void tdm_load_2d_f16(const void* gsrc, unsigned lds_addr,
                                                unsigned td0, unsigned td1,
                                                unsigned t0, unsigned t1,
                                                unsigned stride0) {
  unsigned long long ga = (unsigned long long)(size_t)gsrc;
  u32x4 g0;
  g0[0] = 1u;                                            // count=1, user mode
  g0[1] = lds_addr;                                      // LDS byte address
  g0[2] = (unsigned)ga;                                  // global_addr[31:0]
  g0[3] = ((unsigned)(ga >> 32) & 0x01FFFFFFu) | (2u << 30);  // addr[56:32] | type=2
  i32x8 g1;
  g1[0] = (int)((1u << 16) | (1u << 20) | (3u << 22) | (3u << 25));
          // data_size=2B | pad_enable | pad_interval=16DW | pad_amount=4DW
  g1[1] = (int)((td0 & 0xFFFFu) << 16);                  // barrier_addr=0 | tensor_dim0.lo
  g1[2] = (int)((td0 >> 16) | ((td1 & 0xFFFFu) << 16));  // tensor_dim0.hi | tensor_dim1.lo
  g1[3] = (int)((td1 >> 16) | (t0 << 16));               // tensor_dim1.hi | tile_dim0
  g1[4] = (int)(t1 & 0xFFFFu);                           // tile_dim1 | tile_dim2=0 (2D)
  g1[5] = (int)stride0;                                  // tensor_dim0_stride[31:0]
  g1[6] = 0;                                             // stride0.hi | stride1.lo (unused)
  g1[7] = 0;
  i32x4 z4 = {0, 0, 0, 0};
  i32x8 z8 = {0, 0, 0, 0, 0, 0, 0, 0};
  __builtin_amdgcn_tensor_load_to_lds(g0, g1, z4, z4, z8, 0);
}
#endif

template <int N>
__device__ __forceinline__ void tdm_wait() {
#if HAVE_TDM
#if __has_builtin(__builtin_amdgcn_s_wait_tensorcnt)
  __builtin_amdgcn_s_wait_tensorcnt(N);
#else
  asm volatile("s_wait_tensorcnt %0" :: "n"(N) : "memory");
#endif
#endif
}

// ---------------------------------------------------------------------------
// C[M,256] = Ah[M,K] @ Bt[256,K]^T : f16 operands (pre-converted, K-contiguous),
// f32 accumulate via v_wmma_f32_16x16x32_f16.
// Block: 256 threads = 8 waves; block owns 16 M-rows; wave w owns cols [32w,32w+32).
// Tiles staged to LDS by the Tensor Data Mover, double-buffered, padded rows.
// Requires M%16==0, K%32==0.
// ---------------------------------------------------------------------------
template <int K>
__global__ __launch_bounds__(256)
void gemm_wmma(const f16* __restrict__ Ah, const f16* __restrict__ Bt,
               float* __restrict__ C) {
  __shared__ f16 As[2][16 * TSTR];
  __shared__ f16 Bs[2][HIDF * TSTR];
  const int tid  = threadIdx.x;
  const int wave = tid >> 5;
  const int lane = tid & 31;
  const int m0   = blockIdx.x * 16;
  const int row  = lane & 15;          // A-frag row / B-frag column
  const int kb   = (lane >> 4) * 16;   // A-frag K sub-base
  const int ks   = (lane >> 4) * 16;   // B-frag K sub-base
  const int n0   = wave * 32;

  v8f acc0 = {};
  v8f acc1 = {};

  // ---- tile staging (TDM preferred; manual padded copy as fallback) ----
  auto stage = [&](int k0, int buf) {
#if HAVE_TDM
    if (wave == 0) {
      tdm_load_2d_f16(Ah + (size_t)m0 * K + k0, (unsigned)(size_t)&As[buf][0],
                      (unsigned)(K - k0), 16u, 32u, 16u, (unsigned)K);
      tdm_load_2d_f16(Bt + k0, (unsigned)(size_t)&Bs[buf][0],
                      (unsigned)(K - k0), (unsigned)HIDF, 32u, (unsigned)HIDF,
                      (unsigned)K);
    }
#else
    // B: 1024 x 16B chunks, 4 per thread; A: 64 x 16B chunks (threads < 64)
    for (int it = 0; it < 4; ++it) {
      int s2 = tid + it * 256;
      int n = s2 >> 2, part = s2 & 3;
      *(u32x4*)&Bs[buf][n * TSTR + part * 8] =
          *(const u32x4*)(Bt + (size_t)n * K + k0 + part * 8);
    }
    if (tid < 64) {
      int r = tid >> 2, part = tid & 3;
      *(u32x4*)&As[buf][r * TSTR + part * 8] =
          *(const u32x4*)(Ah + (size_t)(m0 + r) * K + k0 + part * 8);
    }
#endif
  };

  stage(0, 0);
  constexpr int S = K / 32;
#pragma unroll
  for (int s = 0; s < S; ++s) {
    const int cur = s & 1;
    if (s + 1 < S) {          // prefetch next tile into alternate buffer
      stage(32 * (s + 1), cur ^ 1);
      tdm_wait<2>();          // current step's 2 TDM ops retired; next 2 in flight
    } else {
      tdm_wait<0>();
    }
    __syncthreads();

    const f16* Ab = &As[cur][0];
    const f16* Bb = &Bs[cur][0];

    // A fragment: lane<16 holds K {0..15}; lane>=16 holds {16..31}
    v8h alo = *(const v8h*)(Ab + row * TSTR + kb);
    v8h ahi = *(const v8h*)(Ab + row * TSTR + kb + 8);
    v16h a;
#pragma unroll
    for (int i = 0; i < 8; ++i) { a[i] = alo[i]; a[i + 8] = ahi[i]; }

    // B fragment: b[e] = B[ks+e][col] -> contiguous halfs in K-major LDS rows
    v8h b0lo = *(const v8h*)(Bb + (n0 + row) * TSTR + ks);
    v8h b0hi = *(const v8h*)(Bb + (n0 + row) * TSTR + ks + 8);
    v8h b1lo = *(const v8h*)(Bb + (n0 + 16 + row) * TSTR + ks);
    v8h b1hi = *(const v8h*)(Bb + (n0 + 16 + row) * TSTR + ks + 8);
    v16h b0, b1;
#pragma unroll
    for (int i = 0; i < 8; ++i) {
      b0[i] = b0lo[i]; b0[i + 8] = b0hi[i];
      b1[i] = b1lo[i]; b1[i + 8] = b1hi[i];
    }

    acc0 = __builtin_amdgcn_wmma_f32_16x16x32_f16(false, a, false, b0,
                                                  (short)0, acc0, false, false);
    acc1 = __builtin_amdgcn_wmma_f32_16x16x32_f16(false, a, false, b1,
                                                  (short)0, acc1, false, false);
    __syncthreads();
  }

  // D layout: VGPR r, lane l -> M = r + 8*(l>>4), N = l&15
  const int mr = m0 + (lane >> 4) * 8;
  const int nc = n0 + row;
#pragma unroll
  for (int r = 0; r < 8; ++r) {
    C[(size_t)(mr + r) * HIDF + nc]      = acc0[r];
    C[(size_t)(mr + r) * HIDF + nc + 16] = acc1[r];
  }
}

// ---------------------------------------------------------------------------
// Precision prep kernels (one-shot, bandwidth-trivial)
// ---------------------------------------------------------------------------
__global__ void k_cvt_f16(const float* __restrict__ in, f16* __restrict__ out, int n) {
  int i = blockIdx.x * blockDim.x + threadIdx.x;
  if (i < n) out[i] = (f16)in[i];
}

// W[K][256] -> Wt[256][K] in f16 (K-contiguous rows for WMMA B fragments)
__global__ void k_wt_f16(const float* __restrict__ W, f16* __restrict__ Wt, int K) {
  int i = blockIdx.x * blockDim.x + threadIdx.x;
  if (i < K * HIDF) {
    int n = i & (HIDF - 1);
    int k = i >> 8;
    Wt[(size_t)n * K + k] = (f16)W[(size_t)k * HIDF + n];
  }
}

// ---------------------------------------------------------------------------
// Graph kernels (HBM-bandwidth bound)
// ---------------------------------------------------------------------------
__global__ void k_fill_int(int* __restrict__ p, int v, int n) {
  int i = blockIdx.x * blockDim.x + threadIdx.x;
  if (i < n) p[i] = v;
}

__global__ void k_count(const int* __restrict__ idx, int* __restrict__ deg, int n) {
  int i = blockIdx.x * blockDim.x + threadIdx.x;
  if (i < n) atomicAdd(&deg[idx[i]], 1);
}

__global__ void k_rsqrt_int(const int* __restrict__ deg, float* __restrict__ dinv, int n) {
  int i = blockIdx.x * blockDim.x + threadIdx.x;
  if (i < n) dinv[i] = rsqrtf((float)deg[i]);   // deg >= 1 (self loop)
}

// out[i][f] = bias[f] + xw[i][f] * dinv[i]^2   (self-loop term + bias init)
__global__ void k_conv_init(const float* __restrict__ xw, const float* __restrict__ dinv,
                            const float* __restrict__ bias, float* __restrict__ out) {
  int i = blockIdx.x, f = threadIdx.x;
  float d = dinv[i];
  out[(size_t)i * HIDF + f] = bias[f] + xw[(size_t)i * HIDF + f] * d * d;
}

// out[dst[e]][f] += xw[src[e]][f] * dinv[src]*dinv[dst]
__global__ void k_conv_scatter(const float* __restrict__ xw, const float* __restrict__ dinv,
                               const int* __restrict__ src, const int* __restrict__ dst,
                               float* __restrict__ out) {
  int e = blockIdx.x, f = threadIdx.x;
  int r = src[e], c = dst[e];
  float w = dinv[r] * dinv[c];
  atomicAdd(&out[(size_t)c * HIDF + f], xw[(size_t)r * HIDF + f] * w);
}

// train-mode BN stats per feature: one block per column
__global__ void k_bn_stats(const float* __restrict__ h, float* __restrict__ mu,
                           float* __restrict__ rstd, int nrows) {
  __shared__ float s1[256], s2[256];
  int f = blockIdx.x;
  float a = 0.f, b = 0.f;
  for (int i = threadIdx.x; i < nrows; i += blockDim.x) {
    float v = h[(size_t)i * HIDF + f];
    a += v; b += v * v;
  }
  s1[threadIdx.x] = a; s2[threadIdx.x] = b;
  __syncthreads();
  for (int s = 128; s > 0; s >>= 1) {
    if (threadIdx.x < s) { s1[threadIdx.x] += s1[threadIdx.x + s];
                           s2[threadIdx.x] += s2[threadIdx.x + s]; }
    __syncthreads();
  }
  if (threadIdx.x == 0) {
    float m = s1[0] / nrows;
    float v = s2[0] / nrows - m * m;
    mu[f] = m;
    rstd[f] = rsqrtf(v + 1e-5f);
  }
}

// BN + ReLU; also emits an f16 shadow copy (next GEMM's A operand)
__global__ void k_bn_relu(float* __restrict__ h, f16* __restrict__ hh,
                          const float* __restrict__ mu, const float* __restrict__ rstd,
                          const float* __restrict__ gamma, const float* __restrict__ beta,
                          int total) {
  int i = blockIdx.x * blockDim.x + threadIdx.x;
  if (i < total) {
    int f = i & (HIDF - 1);
    float v = (h[i] - mu[f]) * rstd[f] * gamma[f] + beta[f];
    v = v > 0.f ? v : 0.f;
    h[i] = v;
    hh[i] = (f16)v;
  }
}

__global__ void k_relu(float* __restrict__ h, int total) {
  int i = blockIdx.x * blockDim.x + threadIdx.x;
  if (i < total) { float v = h[i]; h[i] = v > 0.f ? v : 0.f; }
}

// Algebraic collapse of the line-graph linear (OUT_F == 1):
// p[n] = dot(h[n], W3[0:256]);  q[n] = dot(h[n], W3[256:512]).  Wave per node.
__global__ __launch_bounds__(256)
void k_pq(const float* __restrict__ h, const float* __restrict__ W3,
          float* __restrict__ p, float* __restrict__ q, int n) {
  int w = (blockIdx.x * blockDim.x + threadIdx.x) >> 5;
  int lane = threadIdx.x & 31;
  if (w >= n) return;
  const float* row = h + (size_t)w * HIDF;
  float sp = 0.f, sq = 0.f;
  for (int f = lane; f < HIDF; f += 32) {
    float v = row[f];
    sp += v * W3[f];
    sq += v * W3[HIDF + f];
  }
  for (int off = 16; off > 0; off >>= 1) {
    sp += __shfl_down(sp, off);
    sq += __shfl_down(sq, off);
  }
  if (lane == 0) { p[w] = sp; q[w] = sq; }
}

// line-node features + self-loop init: xl[e]=p[src]+q[dst]; z[e]=b3+xl*dli^2
__global__ void k_line_init(const float* __restrict__ p, const float* __restrict__ q,
                            const int* __restrict__ src, const int* __restrict__ dst,
                            const int* __restrict__ ldeg, float* __restrict__ dli,
                            float* __restrict__ xl, float* __restrict__ zpre,
                            const float* __restrict__ b3, int E) {
  int i = blockIdx.x * blockDim.x + threadIdx.x;
  if (i < E) {
    float d = rsqrtf((float)ldeg[i]);
    dli[i] = d;
    float v = p[src[i]] + q[dst[i]];
    xl[i] = v;
    zpre[i] = b3[0] + v * d * d;
  }
}

__global__ void k_line_scatter(const float* __restrict__ xl, const float* __restrict__ dli,
                               const int* __restrict__ lr, const int* __restrict__ lc,
                               float* __restrict__ zpre, int L) {
  int i = blockIdx.x * blockDim.x + threadIdx.x;
  if (i < L) {
    int r = lr[i], c = lc[i];
    atomicAdd(&zpre[c], xl[r] * dli[r] * dli[c]);
  }
}

__global__ void k_sigmoid(const float* __restrict__ z, float* __restrict__ out, int n) {
  int i = blockIdx.x * blockDim.x + threadIdx.x;
  if (i < n) out[i] = 1.f / (1.f + expf(-z[i]));
}

// ---------------------------------------------------------------------------
extern "C" void kernel_launch(void* const* d_in, const int* in_sizes, int n_in,
                              void* d_out, int out_size, void* d_ws, size_t ws_size,
                              hipStream_t stream) {
  const float* x     = (const float*)d_in[0];
  const float* W1    = (const float*)d_in[1];
  const float* b1    = (const float*)d_in[2];
  const float* gamma = (const float*)d_in[3];
  const float* beta  = (const float*)d_in[4];
  const float* W2    = (const float*)d_in[5];
  const float* b2    = (const float*)d_in[6];
  const float* W3    = (const float*)d_in[7];
  const float* b3    = (const float*)d_in[8];
  const int*   ei    = (const int*)d_in[9];
  const int*   lei   = (const int*)d_in[10];

  const int N = in_sizes[0] / 128;  // 30000 nodes (multiple of 16)
  const int E = in_sizes[9] / 2;    // 200000 edges
  const int L = in_sizes[10] / 2;   // line-graph edges
  const int* src = ei;
  const int* dst = ei + E;
  const int* lr  = lei;
  const int* lc  = lei + L;

  // workspace carve (256B-aligned slabs; all fully initialized every call)
  char* cur = (char*)d_ws;
  auto alloc = [&](size_t bytes) -> void* {
    void* p = cur;
    cur += (bytes + 255) & ~(size_t)255;
    return p;
  };
  float* xw   = (float*)alloc((size_t)N * HIDF * 4);  // GEMM output
  float* h    = (float*)alloc((size_t)N * HIDF * 4);  // conv output / activations
  f16*   xh   = (f16*)alloc((size_t)N * 128 * 2);     // f16 shadow of x
  f16*   hh   = (f16*)alloc((size_t)N * HIDF * 2);    // f16 shadow of h1
  f16*   w1t  = (f16*)alloc((size_t)128 * HIDF * 2);  // W1^T f16
  f16*   w2t  = (f16*)alloc((size_t)HIDF * HIDF * 2); // W2^T f16
  int*   deg  = (int*)alloc((size_t)N * 4);
  float* dinv = (float*)alloc((size_t)N * 4);
  float* mu   = (float*)alloc(HIDF * 4);
  float* rstd = (float*)alloc(HIDF * 4);
  float* p    = (float*)alloc((size_t)N * 4);
  float* q    = (float*)alloc((size_t)N * 4);
  int*   ldeg = (int*)alloc((size_t)E * 4);
  float* dli  = (float*)alloc((size_t)E * 4);
  float* xl   = (float*)alloc((size_t)E * 4);
  float* zpre = (float*)alloc((size_t)E * 4);

  // precision prep: f16 operands in WMMA-friendly layouts
  k_cvt_f16<<<((size_t)N * 128 + 255) / 256, 256, 0, stream>>>(x, xh, N * 128);
  k_wt_f16<<<(128 * HIDF + 255) / 256, 256, 0, stream>>>(W1, w1t, 128);
  k_wt_f16<<<(HIDF * HIDF + 255) / 256, 256, 0, stream>>>(W2, w2t, HIDF);

  // node-graph normalization: deg = in-degree(dst) + 1 self loop
  k_fill_int<<<(N + 255) / 256, 256, 0, stream>>>(deg, 1, N);
  k_count<<<(E + 255) / 256, 256, 0, stream>>>(dst, deg, E);
  k_rsqrt_int<<<(N + 255) / 256, 256, 0, stream>>>(deg, dinv, N);

  // layer 1: GEMM (TDM + WMMA) -> normalized scatter -> BN(train) -> ReLU
  gemm_wmma<128><<<N / 16, 256, 0, stream>>>(xh, w1t, xw);
  k_conv_init<<<N, 256, 0, stream>>>(xw, dinv, b1, h);
  k_conv_scatter<<<E, 256, 0, stream>>>(xw, dinv, src, dst, h);
  k_bn_stats<<<HIDF, 256, 0, stream>>>(h, mu, rstd, N);
  k_bn_relu<<<((size_t)N * HIDF + 255) / 256, 256, 0, stream>>>(h, hh, mu, rstd, gamma, beta, N * HIDF);

  // layer 2: GEMM (TDM + WMMA) -> normalized scatter -> ReLU
  gemm_wmma<256><<<N / 16, 256, 0, stream>>>(hh, w2t, xw);
  k_conv_init<<<N, 256, 0, stream>>>(xw, dinv, b2, h);
  k_conv_scatter<<<E, 256, 0, stream>>>(xw, dinv, src, dst, h);
  k_relu<<<((size_t)N * HIDF + 255) / 256, 256, 0, stream>>>(h, N * HIDF);

  // line-graph layer: collapse cat(h[src],h[dst]) @ W3 into p[src]+q[dst]
  k_pq<<<((size_t)N * 32 + 255) / 256, 256, 0, stream>>>(h, W3, p, q, N);
  k_fill_int<<<(E + 255) / 256, 256, 0, stream>>>(ldeg, 1, E);
  k_count<<<(L + 255) / 256, 256, 0, stream>>>(lc, ldeg, L);
  k_line_init<<<(E + 255) / 256, 256, 0, stream>>>(p, q, src, dst, ldeg, dli, xl, zpre, b3, E);
  k_line_scatter<<<(L + 255) / 256, 256, 0, stream>>>(xl, dli, lr, lc, zpre, L);
  k_sigmoid<<<(E + 255) / 256, 256, 0, stream>>>(zpre, (float*)d_out, E);
}
